// MACHAttention_59304908423783
// MI455X (gfx1250) — matrix operand order repaired and checked by reading, not compiled, not run on hardware
//
#include <hip/hip_runtime.h>

// ---------------------------------------------------------------------------
// Types for CDNA5 WMMA (wave32): bf16 A/B fragments (16 x bf16), f32 accum (8)
// ---------------------------------------------------------------------------
typedef __attribute__((ext_vector_type(8)))  unsigned short us8;
typedef __attribute__((ext_vector_type(8)))  __bf16         v8bf;
typedef __attribute__((ext_vector_type(16))) __bf16         v16bf;
typedef __attribute__((ext_vector_type(8)))  float          v8f;
typedef __attribute__((ext_vector_type(4)))  int            v4i;

#if defined(__gfx1250__) && __has_builtin(__builtin_amdgcn_global_load_async_to_lds_b128)
#define USE_ASYNC_LDS 1
#else
#define USE_ASYNC_LDS 0
#endif

#if USE_ASYNC_LDS
// async-to-LDS builtin wants (global v4i*, lds v4i*, imm offset, imm cpol)
#define GLOBAL_V4I(p) ((__attribute__((address_space(1))) v4i*)(p))
#define LDS_V4I(p)    ((__attribute__((address_space(3))) v4i*)(p))
#endif

#if defined(__gfx1250__)
template <short N>
static __device__ __forceinline__ void wait_asynccnt() {
#if __has_builtin(__builtin_amdgcn_s_wait_asynccnt)
    __builtin_amdgcn_s_wait_asynccnt(N);
#else
    asm volatile("s_wait_asynccnt %0" ::"i"(N) : "memory");
#endif
}
#else
template <short N>
static __device__ __forceinline__ void wait_asynccnt() {}
#endif

static __device__ __forceinline__ unsigned short f32_to_bf16_rne(float f) {
    unsigned int u = __float_as_uint(f);
    unsigned int lsb = (u >> 16) & 1u;
    u += 0x7fffu + lsb;                    // round-to-nearest-even
    return (unsigned short)(u >> 16);
}

// ---------------------------------------------------------------------------
// fp32 -> bf16 bulk convert
// ---------------------------------------------------------------------------
__global__ __launch_bounds__(256) void cvt_f32_bf16(const float* __restrict__ src,
                                                    unsigned short* __restrict__ dst,
                                                    int count) {
    int i = blockIdx.x * blockDim.x + threadIdx.x;
    int stride = gridDim.x * blockDim.x;
    for (; i < count; i += stride) dst[i] = f32_to_bf16_rne(src[i]);
}

// Of [16][1024][64] (head-major) -> Obf [2048][512] bf16 (token-major, head-interleaved)
__global__ __launch_bounds__(256) void permute_cvt(const float* __restrict__ Of,
                                                   unsigned short* __restrict__ Obf) {
    int g = blockIdx.x * blockDim.x + threadIdx.x;  // < 2048*512
    int j = g & 511, m = g >> 9;
    int b = m >> 10, t = m & 1023, h = j >> 6, d = j & 63;
    Obf[g] = f32_to_bf16_rne(Of[(((b * 8 + h) * 1024) + t) * 64 + d]);
}

// ---------------------------------------------------------------------------
// WMMA GEMM: Y[m][j] = sum_k A[m][k] * W[j][k]   (M=2048, N=512, K=512, bf16)
// Block tile 64(M) x 128(N), 8 waves: wave = (m-sub 0..3, n-sub 0..1),
// each wave: 1 A-frag (16x32) x 4 B-frags (32x16) -> 4 v_wmma_f32_16x16x32_bf16
// Double-buffered LDS tiles staged with GLOBAL_LOAD_ASYNC_TO_LDS_B128
// (ASYNCcnt-tracked; each thread owns 3 async b128 per tile, so
// s_wait_asynccnt 3 after issuing tile i+1 guarantees tile i has landed).
// HEADOUT: scatter to [N=B*H][T][64] layout; else token-major + bias.
// ---------------------------------------------------------------------------
template <bool HEADOUT>
__global__ __launch_bounds__(256) void gemm_wmma(const unsigned short* __restrict__ A,
                                                 const unsigned short* __restrict__ W,
                                                 const float* __restrict__ bias,
                                                 float* __restrict__ out) {
    __shared__ unsigned short As[2][64][32];    // [buf][m][k]  2 x 4 KB
    __shared__ unsigned short Bs[2][128][32];   // [buf][n][k]  2 x 8 KB

    const int tid  = threadIdx.x;
    const int lane = tid & 31;
    const int wave = tid >> 5;
    const int mw = (wave & 3) * 16;          // wave's M offset in block tile
    const int nw = (wave >> 2) * 64;         // wave's N offset in block tile
    const int mTile = (blockIdx.x & 31) * 64;   // 2048/64 = 32 tiles
    const int nTile = (blockIdx.x >> 5) * 128;  // 512/128 = 4  tiles

    v8f acc[4] = {};

    // staging tasks: A = 64x32 bf16 (1 b128/thread), B = 128x32 bf16 (2 b128/thread)
    const int aRow = tid >> 2;               // 0..63
    const int aSeg = (tid & 3) * 8;          // 0,8,16,24
    const int bRow = tid >> 1;               // 0..127
    const int bSeg = (tid & 1) * 16;         // 0,16
    const unsigned short* aG = A + (size_t)(mTile + aRow) * 512 + aSeg;
    const unsigned short* bG = W + (size_t)(nTile + bRow) * 512 + bSeg;

    auto stage = [&](int buf, int k0) {
#if USE_ASYNC_LDS
        __builtin_amdgcn_global_load_async_to_lds_b128(
            GLOBAL_V4I(aG + k0), LDS_V4I(&As[buf][aRow][aSeg]), 0, 0);
        __builtin_amdgcn_global_load_async_to_lds_b128(
            GLOBAL_V4I(bG + k0), LDS_V4I(&Bs[buf][bRow][bSeg]), 0, 0);
        __builtin_amdgcn_global_load_async_to_lds_b128(
            GLOBAL_V4I(bG + k0 + 8), LDS_V4I(&Bs[buf][bRow][bSeg + 8]), 0, 0);
#else
        *(us8*)&As[buf][aRow][aSeg]     = *(const us8*)(aG + k0);
        *(us8*)&Bs[buf][bRow][bSeg]     = *(const us8*)(bG + k0);
        *(us8*)&Bs[buf][bRow][bSeg + 8] = *(const us8*)(bG + k0 + 8);
        if (k0 + 32 < 512) __builtin_prefetch(aG + k0 + 32, 0, 0);
#endif
    };

    stage(0, 0);

    for (int iter = 0; iter < 16; ++iter) {
        const int k0  = iter * 32;
        const int buf = iter & 1;
        if (iter + 1 < 16) stage(buf ^ 1, k0 + 32);
#if USE_ASYNC_LDS
        if (iter + 1 < 16) wait_asynccnt<3>();   // tile `iter` complete (in-order retire)
        else               wait_asynccnt<0>();
#endif
        __syncthreads();   // all waves' tile-`iter` data visible in LDS

        // ---- A fragment: lane<16 -> M=lane, K in {0..7,16..23}; lane>=16 -> K {8..15,24..31}
        const int am = mw + (lane & 15);
        const int ab = (lane < 16) ? 0 : 8;
        union { us8 u; v8bf v; } alo, ahi;
        alo.u = *(const us8*)&As[buf][am][ab];
        ahi.u = *(const us8*)&As[buf][am][ab + 16];
        v16bf afrag;
#pragma unroll
        for (int i = 0; i < 8; ++i) { afrag[i] = alo.v[i]; afrag[i + 8] = ahi.v[i]; }

        // ---- B fragments: lane<16 -> N=lane, K in {0..7,16..23}; lane>=16 -> K {8..15,24..31}
        const int kb = (lane < 16) ? 0 : 8;
#pragma unroll
        for (int f = 0; f < 4; ++f) {
            const int bn = nw + f * 16 + (lane & 15);
            union { us8 u; v8bf v; } blo, bhi;
            blo.u = *(const us8*)&Bs[buf][bn][kb];
            bhi.u = *(const us8*)&Bs[buf][bn][kb + 16];
            v16bf bfrag;
#pragma unroll
            for (int i = 0; i < 8; ++i) { bfrag[i] = blo.v[i]; bfrag[i + 8] = bhi.v[i]; }
            acc[f] = __builtin_amdgcn_wmma_f32_16x16x32_bf16(
                false, afrag, false, bfrag, (short)0, acc[f], false, false);
        }
        __syncthreads();   // done reading buf; next-next stage may overwrite it
    }

    // ---- store: C/D layout: VGPR r -> M = r + (lane<16?0:8), N = lane&15
    const int rowOff = (lane < 16) ? 0 : 8;
    const int nlane  = lane & 15;
#pragma unroll
    for (int f = 0; f < 4; ++f) {
        const int gj = nTile + nw + f * 16 + nlane;
#pragma unroll
        for (int r = 0; r < 8; ++r) {
            const int gm = mTile + mw + r + rowOff;
            float v = acc[f][r];
            if (HEADOUT) {
                int b = gm >> 10, t = gm & 1023, h = gj >> 6, d = gj & 63;
                out[(((size_t)(b * 8 + h) * 1024) + t) * 64 + d] = v;
            } else {
                out[(size_t)gm * 512 + gj] = v + bias[gj];
            }
        }
    }
}

// ---------------------------------------------------------------------------
// LSH bucket assignment: one thread per (n,t). proj = K . planes_T, tanh/scale,
// logits vs hypercube corners, per-table softmax, top-2, renormalize.
// ---------------------------------------------------------------------------
__global__ __launch_bounds__(256) void lsh_assign(const float* __restrict__ Kf,        // [16][1024][64]
                                                  const float* __restrict__ planesT,   // [64][12]
                                                  const float* __restrict__ protosT,   // [3][8]
                                                  const float* __restrict__ logit_temp,
                                                  float* __restrict__ assign) {        // [16][1024][32]
    int g = blockIdx.x * blockDim.x + threadIdx.x;  // < 16384
    const float* kvec = Kf + (size_t)g * 64;
    float scale = __expf(logit_temp[0]);
    scale = fminf(fmaxf(scale, 0.01f), 20.0f);
    float inv_scale = 1.0f / scale;

    float proj[12];
#pragma unroll
    for (int j = 0; j < 12; ++j) proj[j] = 0.f;
    for (int d = 0; d < 64; ++d) {
        float kd = kvec[d];
#pragma unroll
        for (int j = 0; j < 12; ++j) proj[j] += kd * planesT[d * 12 + j];
    }

    float outv[32];
#pragma unroll
    for (int l = 0; l < 4; ++l) {
        float th0 = tanhf(proj[l * 3 + 0]) * inv_scale;
        float th1 = tanhf(proj[l * 3 + 1]) * inv_scale;
        float th2 = tanhf(proj[l * 3 + 2]) * inv_scale;
        float p[8];
        float mx = -3.4e38f;
#pragma unroll
        for (int r = 0; r < 8; ++r) {
            float s = th0 * protosT[r] + th1 * protosT[8 + r] + th2 * protosT[16 + r];
            p[r] = s;
            mx = fmaxf(mx, s);
        }
        float sum = 0.f;
#pragma unroll
        for (int r = 0; r < 8; ++r) { p[r] = __expf(p[r] - mx); sum += p[r]; }
        float inv = 1.0f / sum;
#pragma unroll
        for (int r = 0; r < 8; ++r) p[r] *= inv;
        // top-2 (first occurrence wins ties, matching lax.top_k)
        int i1 = 0;
#pragma unroll
        for (int r = 1; r < 8; ++r) if (p[r] > p[i1]) i1 = r;
        int i2 = (i1 == 0) ? 1 : 0;
#pragma unroll
        for (int r = 0; r < 8; ++r) if (r != i1 && p[r] > p[i2]) i2 = r;
        float dn = p[i1] + p[i2] + 1e-6f;
#pragma unroll
        for (int r = 0; r < 8; ++r) outv[l * 8 + r] = 0.f;
        outv[l * 8 + i1] = p[i1] / dn;
        outv[l * 8 + i2] = p[i2] / dn;
    }
    float* dst = assign + (size_t)g * 32;
#pragma unroll
    for (int s = 0; s < 32; ++s) dst[s] = outv[s];
}

// ---------------------------------------------------------------------------
// Fused causal bucket scan + bucket attention. One block per head n.
// LDS holds running numK/numV [32][64] (padded stride 68 to dodge bank
// conflicts) and denom [32]. score_s = (Q . numK_s) / (den_s+eps) / 8;
// out = sum_s softmax(score)_s/(den_s+eps) * numV_s.
// ---------------------------------------------------------------------------
__global__ __launch_bounds__(256) void bucket_scan(const float* __restrict__ Qf,
                                                   const float* __restrict__ Kf,
                                                   const float* __restrict__ Vf,
                                                   const float* __restrict__ assign,
                                                   float* __restrict__ Of) {
    __shared__ float numK[32][68];
    __shared__ float numV[32][68];
    __shared__ float den[32];
    __shared__ float asg[32];
    __shared__ float Kt[64], Vt[64], Qt[64];
    __shared__ float sc[32];
    __shared__ float wgt[32];

    const int n = blockIdx.x;
    const int tid = threadIdx.x;

    for (int i = tid; i < 32 * 68; i += 256) {
        ((float*)numK)[i] = 0.f;
        ((float*)numV)[i] = 0.f;
    }
    if (tid < 32) den[tid] = 0.f;
    __syncthreads();

    const int s8 = tid >> 3;        // bucket 0..31
    const int d8 = (tid & 7) * 8;   // dim chunk base

    for (int t = 0; t < 1024; ++t) {
        const size_t base = ((size_t)n * 1024 + t) * 64;
        if (tid < 64) {
            Kt[tid] = Kf[base + tid];
            Vt[tid] = Vf[base + tid];
            Qt[tid] = Qf[base + tid];
        } else if (tid < 96) {
            asg[tid - 64] = assign[((size_t)n * 1024 + t) * 32 + (tid - 64)];
        }
        __syncthreads();

        // inclusive prefix update (top-2 sparsity: most groups skip)
        {
            float a = asg[s8];
            if (a != 0.f) {
#pragma unroll
                for (int j = 0; j < 8; ++j) {
                    int d = d8 + j;
                    numK[s8][d] += a * Kt[d];
                    numV[s8][d] += a * Vt[d];
                }
                if ((tid & 7) == 0) den[s8] += a;
            }
        }
        __syncthreads();

        // scores: 8-lane partial dot + intra-wave xor reduce
        {
            float part = 0.f;
#pragma unroll
            for (int j = 0; j < 8; ++j) {
                int d = d8 + j;
                part += Qt[d] * numK[s8][d];
            }
            part += __shfl_xor(part, 1);
            part += __shfl_xor(part, 2);
            part += __shfl_xor(part, 4);
            if ((tid & 7) == 0) sc[s8] = part / (den[s8] + 1e-6f) * 0.125f;  // 1/sqrt(64)
        }
        __syncthreads();

        // softmax over 32 buckets in wave 0 (all 32 lanes active)
        if (tid < 32) {
            float v = sc[tid];
            float m = v;
            m = fmaxf(m, __shfl_xor(m, 1));
            m = fmaxf(m, __shfl_xor(m, 2));
            m = fmaxf(m, __shfl_xor(m, 4));
            m = fmaxf(m, __shfl_xor(m, 8));
            m = fmaxf(m, __shfl_xor(m, 16));
            float e = __expf(v - m);
            float sum = e;
            sum += __shfl_xor(sum, 1);
            sum += __shfl_xor(sum, 2);
            sum += __shfl_xor(sum, 4);
            sum += __shfl_xor(sum, 8);
            sum += __shfl_xor(sum, 16);
            wgt[tid] = (e / sum) / (den[tid] + 1e-6f);
        }
        __syncthreads();

        // output: 64 dims, each sums over 32 buckets
        if (tid < 64) {
            float o = 0.f;
#pragma unroll
            for (int s = 0; s < 32; ++s) o += wgt[s] * numV[s][tid];
            Of[base + tid] = o;
        }
        __syncthreads();
    }
}

// ---------------------------------------------------------------------------
// Launch
// ---------------------------------------------------------------------------
extern "C" void kernel_launch(void* const* d_in, const int* in_sizes, int n_in,
                              void* d_out, int out_size, void* d_ws, size_t ws_size,
                              hipStream_t stream) {
    (void)in_sizes; (void)n_in; (void)out_size; (void)ws_size;

    const float* x          = (const float*)d_in[0];  // [2,1024,512]
    const float* Wq         = (const float*)d_in[1];  // [512,512]
    const float* Wk         = (const float*)d_in[2];
    const float* Wv         = (const float*)d_in[3];
    const float* Wout       = (const float*)d_in[4];
    const float* bout       = (const float*)d_in[5];  // [512]
    const float* logit_temp = (const float*)d_in[6];  // [1]
    const float* planesT    = (const float*)d_in[7];  // [64,12]
    const float* protosT    = (const float*)d_in[8];  // [3,8]
    float* out = (float*)d_out;                       // [2,1024,512]

    char* ws = (char*)d_ws;
    size_t off = 0;
    auto alloc = [&](size_t bytes) {
        void* p = ws + off;
        off += (bytes + 255) & ~(size_t)255;
        return p;
    };
    unsigned short* xbf    = (unsigned short*)alloc((size_t)2048 * 512 * 2);
    unsigned short* wqbf   = (unsigned short*)alloc((size_t)512 * 512 * 2);
    unsigned short* wkbf   = (unsigned short*)alloc((size_t)512 * 512 * 2);
    unsigned short* wvbf   = (unsigned short*)alloc((size_t)512 * 512 * 2);
    unsigned short* woutbf = (unsigned short*)alloc((size_t)512 * 512 * 2);
    float* Qf     = (float*)alloc((size_t)16 * 1024 * 64 * 4);
    float* Kf     = (float*)alloc((size_t)16 * 1024 * 64 * 4);
    float* Vf     = (float*)alloc((size_t)16 * 1024 * 64 * 4);
    float* assign = (float*)alloc((size_t)16 * 1024 * 32 * 4);
    float* Of     = (float*)alloc((size_t)16 * 1024 * 64 * 4);
    unsigned short* obf = (unsigned short*)alloc((size_t)2048 * 512 * 2);

    cvt_f32_bf16<<<256, 256, 0, stream>>>(x, xbf, 2048 * 512);
    cvt_f32_bf16<<<64, 256, 0, stream>>>(Wq, wqbf, 512 * 512);
    cvt_f32_bf16<<<64, 256, 0, stream>>>(Wk, wkbf, 512 * 512);
    cvt_f32_bf16<<<64, 256, 0, stream>>>(Wv, wvbf, 512 * 512);
    cvt_f32_bf16<<<64, 256, 0, stream>>>(Wout, woutbf, 512 * 512);

    gemm_wmma<true><<<128, 256, 0, stream>>>(xbf, wqbf, nullptr, Qf);
    gemm_wmma<true><<<128, 256, 0, stream>>>(xbf, wkbf, nullptr, Kf);
    gemm_wmma<true><<<128, 256, 0, stream>>>(xbf, wvbf, nullptr, Vf);

    lsh_assign<<<64, 256, 0, stream>>>(Kf, planesT, protosT, logit_temp, assign);
    bucket_scan<<<16, 256, 0, stream>>>(Qf, Kf, Vf, assign, Of);

    permute_cvt<<<4096, 256, 0, stream>>>(Of, obf);
    gemm_wmma<false><<<128, 256, 0, stream>>>(obf, woutbf, bout, out);
}
